// BiLSTMLag1_15779709845824
// MI455X (gfx1250) — matrix operand order repaired and checked by reading, not compiled, not run on hardware
//
#include <hip/hip_runtime.h>

// BiLSTM (4 stacked bidirectional layers) for MI455X / gfx1250.
// grid = 64 WGs (one per 16-batch tile), block = 64 threads = 2 wave32s
// (wave0 = forward, wave1 = backward). Gate GEMMs: v_wmma_f32_16x16x32_f16,
// M=16 batch tile, N=4H padded to 16, K padded to 32. Weight B-fragments are
// packed once per layer and held in VGPRs across the whole time loop; x
// operands are software-pipelined (loads for t+1 issued during step t); the
// recurrent h operand bounces through per-wave-private LDS with wave-local
// ordering only (no workgroup barrier inside the time loop). Activations are
// branch-free v_exp_f32/v_rcp_f32 sequences. Inter-layer activation sequences
// ping-pong through d_ws in f16 ([B][T][48]-half layout, L2-resident).

typedef _Float16 half_t;
typedef __attribute__((ext_vector_type(16))) _Float16 v16h;
typedef __attribute__((ext_vector_type(8)))  float    v8f;

#define T_SEQ   1024
#define B_TOT   1024
#define TILE_B  16
#define N_WGS   64
#define FEAT    48    // padded feature stride (halves) in seq buffers
#define GSTRIDE 80    // gate scratch row stride (floats), max 4H = 80

struct Args {
  const float* x;
  const float* w_ih[4][2];
  const float* w_hh[4][2];
  const float* b_ih[4][2];
  const float* b_hh[4][2];
  const float* fc_w;
  const float* fc_b;
  float*  out;
  half_t* buf0;
  half_t* buf1;
};

__device__ __forceinline__ float sigf(float x) {
  return __builtin_amdgcn_rcpf(1.f + __expf(-x));       // v_exp_f32 + v_rcp_f32
}
__device__ __forceinline__ float tanhfast(float x) {
  return 2.f * __builtin_amdgcn_rcpf(1.f + __expf(-2.f * x)) - 1.f;  // 2*sig(2x)-1
}
// Wave-local LDS ordering: same-wave LDS ops complete in order, so a
// zero-instruction scheduling barrier + wavefront fence is sufficient for the
// per-wave-private s_gates / s_h regions.
__device__ __forceinline__ void wave_lds_sync() {
  __builtin_amdgcn_fence(__ATOMIC_RELEASE, "wavefront");
  __builtin_amdgcn_wave_barrier();
  __builtin_amdgcn_fence(__ATOMIC_ACQUIRE, "wavefront");
}

__global__ __launch_bounds__(64) void bilstm4_kernel(Args args) {
  const int lane = threadIdx.x & 31;
  const int dir  = threadIdx.x >> 5;           // 0 = forward wave, 1 = backward wave
  const long bBase = (long)blockIdx.x * TILE_B;

  __shared__ half_t s_wfrag[2][7680];          // staging for B fragments (hoisted to VGPRs)
  __shared__ float  s_gates[2][16 * GSTRIDE];  // C-fragment spill for elementwise phase
  __shared__ half_t s_h[2][16 * 32];           // recurrent h in A-fragment row-major layout
  __shared__ float  s_bias[2][80];             // b_ih + b_hh

  half_t* bufs0 = args.buf0;
  half_t* bufs1 = args.buf1;

  // ---- phase 0: pack layer-1 input (x_t, x_{t-1}) for this batch tile ----
  for (int it = threadIdx.x; it < TILE_B * T_SEQ; it += 64) {
    const int m = it & (TILE_B - 1);
    const int t = it >> 4;
    const long b = bBase + m;
    const float* xp = args.x + (b * T_SEQ + t) * 8;
    half_t* op = bufs0 + (b * T_SEQ + t) * FEAT;
#pragma unroll
    for (int f = 0; f < 8; ++f) op[f] = (half_t)xp[f];
    if (t > 0) {
      const float* xl = xp - 8;
#pragma unroll
      for (int f = 0; f < 8; ++f) op[8 + f] = (half_t)xl[f];
    } else {
#pragma unroll
      for (int f = 0; f < 8; ++f) op[8 + f] = (half_t)0.f;
    }
    // feats 16..47 may hold garbage: matching weight rows are zero-padded,
    // so garbage * 0 contributes nothing (deterministic).
  }
  __syncthreads();

  const int row  = lane & 15;                  // A-fragment row (batch in tile)
  const int ksel = (lane & 16) ? 8 : 0;        // A-fragment K base select
  const int bsel = (lane & 16) ? 16 : 0;       // unused; B packing uses ksel (same k order as A)
  const int mb   = (lane & 16) ? 8 : 0;        // C-fragment M base

  const int dinA[4] = {16, 40, 40, 20};
  const int HA[4]   = {20, 20, 10, 10};
  const int KXA[4]  = {1, 2, 2, 1};            // K-steps of 32 for the x operand
  const int NTA[4]  = {5, 5, 3, 3};            // N tiles of 16 covering 4H (padded)

  float creg[10];                              // cell state, <= 16*H/32 values per lane

#pragma unroll
  for (int l = 0; l < 4; ++l) {
    const int din = dinA[l], H = HA[l], kxs = KXA[l], nt = NTA[l];
    const int fourH = 4 * H;
    const half_t* inb = (l & 1) ? bufs1 : bufs0;
    half_t* outb      = (l & 1) ? bufs0 : bufs1;

    // ---- pack this layer's weights into WMMA B-fragment layout (own dir) ----
    const float* Wih = args.w_ih[l][dir];
    const float* Whh = args.w_hh[l][dir];
    for (int j = 0; j < nt; ++j) {
      for (int ks = 0; ks <= kxs; ++ks) {
        const bool isH  = (ks == kxs);         // last K-step slot holds W_hh
        const float* W  = isH ? Whh : Wih;
        const int kdim  = isH ? H : din;
        half_t* frag = &s_wfrag[dir][((j * (kxs + 1) + ks) * 32 + lane) * 16];
        const int n = j * 16 + (lane & 15);
#pragma unroll
        for (int e = 0; e < 16; ++e) {
          const int k = (isH ? 0 : ks * 32) + ksel + e + (e >= 8 ? 8 : 0);
          const float w = (k < kdim && n < fourH) ? W[n * kdim + k] : 0.f;
          frag[e] = (half_t)w;
        }
      }
    }
    for (int n = lane; n < fourH; n += 32)
      s_bias[dir][n] = args.b_ih[l][dir][n] + args.b_hh[l][dir][n];
    for (int i = lane; i < 16 * 32; i += 32) s_h[dir][i] = (half_t)0.f;
#pragma unroll
    for (int i = 0; i < 10; ++i) creg[i] = 0.f;
    __syncthreads();

    // ---- hoist this layer's B fragments into VGPRs (constant over T) ----
    v16h bfr[15];                              // nt*(kxs+1) <= 15 fragments used
#pragma unroll
    for (int j = 0; j < 5; ++j) {
#pragma unroll
      for (int ks = 0; ks < 3; ++ks) {
        if (j < nt && ks <= kxs) {
          const half_t* fr = &s_wfrag[dir][((j * (kxs + 1) + ks) * 32 + lane) * 16];
          v16h b;
#pragma unroll
          for (int e = 0; e < 16; ++e) b[e] = fr[e];
          bfr[j * (kxs + 1) + ks] = b;
        }
      }
    }

    const int items = (TILE_B * H) / 32;       // 10 (H=20) or 5 (H=10)

    // ---- hoist per-item bias values + output pointers into registers ----
    const int dstep = dir ? -1 : 1;
    const int t0    = dir ? (T_SEQ - 1) : 0;
    const long xStep = (long)dstep * FEAT;
    float bzi[10], bzf[10], bzg[10], bzo[10];
    half_t* optr[10];
#pragma unroll
    for (int it = 0; it < 10; ++it) {
      if (it < items) {
        const int idx = it * 32 + lane;
        const int m = idx / H;
        const int k = idx - m * H;
        bzi[it] = s_bias[dir][k];
        bzf[it] = s_bias[dir][H + k];
        bzg[it] = s_bias[dir][2 * H + k];
        bzo[it] = s_bias[dir][3 * H + k];
        optr[it] = outb + ((bBase + m) * T_SEQ + t0) * FEAT + (dir ? H + k : k);
      }
    }

    // ---- prime the software-pipelined x operand ----
    const half_t* xrow = inb + ((bBase + row) * T_SEQ + t0) * FEAT;
    v16h axc[2], axn[2];
#pragma unroll
    for (int ks = 0; ks < 2; ++ks) {
      if (ks < kxs) {
        const int k0 = ks * 32 + ksel;
#pragma unroll
        for (int e = 0; e < 8; ++e) axc[ks][e] = xrow[k0 + e];
#pragma unroll
        for (int e = 0; e < 8; ++e) axc[ks][8 + e] = xrow[k0 + 16 + e];
      }
    }

    for (int tt = 0; tt < T_SEQ; ++tt) {
      // issue next step's x loads now; consumed next iteration
      const half_t* xnext = (tt + 1 < T_SEQ) ? (xrow + xStep) : xrow;
#pragma unroll
      for (int ks = 0; ks < 2; ++ks) {
        if (ks < kxs) {
          const int k0 = ks * 32 + ksel;
#pragma unroll
          for (int e = 0; e < 8; ++e) axn[ks][e] = xnext[k0 + e];
#pragma unroll
          for (int e = 0; e < 8; ++e) axn[ks][8 + e] = xnext[k0 + 16 + e];
        }
      }

      // h operand from per-wave LDS (written by previous iteration, same wave)
      v16h ah;
#pragma unroll
      for (int e = 0; e < 8; ++e) ah[e] = s_h[dir][row * 32 + ksel + e];
#pragma unroll
      for (int e = 0; e < 8; ++e) ah[8 + e] = s_h[dir][row * 32 + ksel + 16 + e];

      // gates[16 x 4H] = X_t @ Wih^T + h @ Whh^T  (weights resident in VGPRs)
#pragma unroll
      for (int j = 0; j < 5; ++j) {
        if (j < nt) {
          v8f acc = {0.f, 0.f, 0.f, 0.f, 0.f, 0.f, 0.f, 0.f};
#pragma unroll
          for (int ks = 0; ks < 2; ++ks) {
            if (ks < kxs) {
              acc = __builtin_amdgcn_wmma_f32_16x16x32_f16(
                  false, axc[ks], false, bfr[j * (kxs + 1) + ks],
                  (short)0, acc, false, false);
            }
          }
          acc = __builtin_amdgcn_wmma_f32_16x16x32_f16(
              false, ah, false, bfr[j * (kxs + 1) + kxs],
              (short)0, acc, false, false);
          // spill C fragment (lane = col n, 8 rows) for cross-lane gather
          const int n = j * 16 + (lane & 15);
#pragma unroll
          for (int r = 0; r < 8; ++r) s_gates[dir][(mb + r) * GSTRIDE + n] = acc[r];
        }
      }
      wave_lds_sync();   // per-wave-private region: wave-local ordering suffices

      // ---- elementwise LSTM cell update (branch-free activations) ----
#pragma unroll
      for (int it = 0; it < 10; ++it) {
        if (it < items) {
          const int idx = it * 32 + lane;
          const int m = idx / H;
          const int k = idx - m * H;
          const float* g = &s_gates[dir][m * GSTRIDE];
          const float gi = g[k]         + bzi[it];
          const float gf = g[H + k]     + bzf[it];
          const float gg = g[2 * H + k] + bzg[it];
          const float go = g[3 * H + k] + bzo[it];
          const float c  = sigf(gf) * creg[it] + sigf(gi) * tanhfast(gg);
          const float h  = sigf(go) * tanhfast(c);
          creg[it] = c;
          s_h[dir][m * 32 + k] = (half_t)h;
          *optr[it] = (half_t)h;
          optr[it] += xStep;
        }
      }
      wave_lds_sync();

      xrow += xStep;
#pragma unroll
      for (int ks = 0; ks < 2; ++ks)
        if (ks < kxs) axc[ks] = axn[ks];
    }
    __syncthreads();   // layer boundary: both directions done before next layer reads
  }

  // ---- final FC + sigmoid on h[:, T-1, :20] (layer-4 output is in buf0) ----
  if (threadIdx.x < TILE_B) {
    const long b = bBase + threadIdx.x;
    const half_t* hp = bufs0 + (b * T_SEQ + (T_SEQ - 1)) * FEAT;
    float a = args.fc_b[0];
#pragma unroll
    for (int k = 0; k < 20; ++k) a += (float)hp[k] * args.fc_w[k];
    args.out[b] = sigf(a);
  }
}

extern "C" void kernel_launch(void* const* d_in, const int* in_sizes, int n_in,
                              void* d_out, int out_size, void* d_ws, size_t ws_size,
                              hipStream_t stream) {
  (void)in_sizes; (void)n_in; (void)out_size; (void)ws_size;
  Args a;
  a.x = (const float*)d_in[0];
  for (int l = 0; l < 4; ++l)
    for (int d = 0; d < 2; ++d) {
      const int base = 1 + l * 8 + d * 4;   // dict order: w_ih, w_hh, b_ih, b_hh
      a.w_ih[l][d] = (const float*)d_in[base + 0];
      a.w_hh[l][d] = (const float*)d_in[base + 1];
      a.b_ih[l][d] = (const float*)d_in[base + 2];
      a.b_hh[l][d] = (const float*)d_in[base + 3];
    }
  a.fc_w = (const float*)d_in[33];
  a.fc_b = (const float*)d_in[34];
  a.out  = (float*)d_out;
  const size_t bufElems = (size_t)B_TOT * T_SEQ * FEAT + 64;  // +64 halves slack for padded K reads
  a.buf0 = (half_t*)d_ws;
  a.buf1 = (half_t*)d_ws + bufElems;
  bilstm4_kernel<<<N_WGS, 64, 0, stream>>>(a);
}
